// ShiftAwareAttention_AllLinks_57183194579325
// MI455X (gfx1250) — matrix operand order repaired and checked by reading, not compiled, not run on hardware
//
#include <hip/hip_runtime.h>
#include <hip/hip_bf16.h>

// ---------------------------------------------------------------------------
// ShiftAwareAttention on gfx1250: bf16 WMMA (v_wmma_f32_16x16x32_bf16) path.
// x: [2,320,207,64] f32, w: scalar. Whole working set (~28MB) lives in the
// 192MB L2 -> compute bound; both einsums run on the matrix pipes with f32
// accumulation. 32x32 register tiles double flop/byte vs 16x16.
// ---------------------------------------------------------------------------

typedef __attribute__((ext_vector_type(16))) __bf16 v16bf;
typedef __attribute__((ext_vector_type(8)))  float  v8f;
typedef int v4i __attribute__((vector_size(16)));   // matches async-LDS builtin param

#define cB 2
#define cT 320
#define cL 207
#define cD 64
#define cS 21
#define cSL   (cS * cL)   // 4347
#define cSLP  4352        // padded attn row (multiple of 32 -> aligned 16B frags)
#define cLT32 7           // ceil(207/32)

#define AS1 __attribute__((address_space(1)))
#define AS3 __attribute__((address_space(3)))

// gfx1250 async global->LDS copy (ASYNCcnt-tracked), gated on availability so
// the fallback register-staging path keeps host pass / older toolchains clean.
#if defined(__has_builtin)
# if __has_builtin(__builtin_amdgcn_global_load_async_to_lds_b128) && \
     __has_builtin(__builtin_amdgcn_s_wait_asynccnt)
#  define HAVE_ASYNC_LDS 1
# endif
#endif
#ifndef HAVE_ASYNC_LDS
# define HAVE_ASYNC_LDS 0
#endif

__device__ __forceinline__ int shift_p(int s) { return (s == 0) ? 287 : (21 - s); }

// --- Kernel 0: f32 -> bf16 convert of x, zero padded attn buffer -----------
__global__ __launch_bounds__(256)
void convert_zero_kernel(const float* __restrict__ x, __bf16* __restrict__ xbf,
                         __bf16* __restrict__ attn_pad,
                         long long nconv, long long npad) {
    long long i = (long long)blockIdx.x * blockDim.x + threadIdx.x;
    if (i < nconv) xbf[i] = (__bf16)x[i];
    if (i < npad)  attn_pad[i] = (__bf16)0.0f;
}

// --- Kernel 1: sim[b,s,l,m] = (w/T) * sum_{t,d} x[b,t,l,d] * x[b,t+p,m,d] ---
// One wave per 32x32 output tile (2x2 WMMA accumulators, A/B fragments each
// reused twice). K = (t,d) contracted 32 at a time.
__global__ __launch_bounds__(32)
void sim_wmma_kernel(const __bf16* __restrict__ xbf, const float* __restrict__ w,
                     float* __restrict__ sim) {
    const int tile = blockIdx.x;            // 0..48 (7x7)
    const int s    = blockIdx.y;            // 0..20
    const int b    = blockIdx.z;            // 0..1
    const int l0   = (tile % cLT32) * 32;   // A rows (target link l)
    const int m0   = (tile / cLT32) * 32;   // B cols (shifted link m)
    const int p    = shift_p(s);
    const int Teff = cT - p;

    const int lane = threadIdx.x;
    const int lr   = lane & 15;
    const int half = lane >> 4;

    const int lrow0 = (l0 + lr      < cL) ? (l0 + lr)      : (cL - 1);
    const int lrow1 = (l0 + 16 + lr < cL) ? (l0 + 16 + lr) : (cL - 1);
    const int mrow0 = (m0 + lr      < cL) ? (m0 + lr)      : (cL - 1);
    const int mrow1 = (m0 + 16 + lr < cL) ? (m0 + 16 + lr) : (cL - 1);

    const size_t bbase   = (size_t)b * cT * cL * cD;
    const size_t tstride = (size_t)cL * cD;
    const size_t aoff0 = bbase + (size_t)lrow0 * cD;
    const size_t aoff1 = bbase + (size_t)lrow1 * cD;
    const size_t boff0 = bbase + (size_t)p * tstride + (size_t)mrow0 * cD;
    const size_t boff1 = bbase + (size_t)p * tstride + (size_t)mrow1 * cD;

    v8f acc[2][2] = {};
    for (int t = 0; t < Teff; ++t) {
        const __bf16* at = xbf + (size_t)t * tstride;
#pragma unroll
        for (int db = 0; db < 2; ++db) {
            const int dbase = db * 32;
            union U { v16bf v; uint4 u[2]; } a0, a1, b0, b1;
            // A 16x32 bf16 layout: lane holds M=lane%16, K = half*8+0..7 and 16+half*8+0..7
            a0.u[0] = *(const uint4*)(at + aoff0 + dbase + half * 8);
            a0.u[1] = *(const uint4*)(at + aoff0 + dbase + 16 + half * 8);
            a1.u[0] = *(const uint4*)(at + aoff1 + dbase + half * 8);
            a1.u[1] = *(const uint4*)(at + aoff1 + dbase + 16 + half * 8);
            // B 32x16 bf16 layout: lane holds N=lane%16, K = half*8+0..7 and 16+half*8+0..7
            b0.u[0] = *(const uint4*)(at + boff0 + dbase + half * 8);
            b0.u[1] = *(const uint4*)(at + boff0 + dbase + 16 + half * 8);
            b1.u[0] = *(const uint4*)(at + boff1 + dbase + half * 8);
            b1.u[1] = *(const uint4*)(at + boff1 + dbase + 16 + half * 8);
            acc[0][0] = __builtin_amdgcn_wmma_f32_16x16x32_bf16(
                false, a0.v, false, b0.v, (short)0, acc[0][0], false, false);
            acc[0][1] = __builtin_amdgcn_wmma_f32_16x16x32_bf16(
                false, a0.v, false, b1.v, (short)0, acc[0][1], false, false);
            acc[1][0] = __builtin_amdgcn_wmma_f32_16x16x32_bf16(
                false, a1.v, false, b0.v, (short)0, acc[1][0], false, false);
            acc[1][1] = __builtin_amdgcn_wmma_f32_16x16x32_bf16(
                false, a1.v, false, b1.v, (short)0, acc[1][1], false, false);
        }
    }

    const float scale = w[0] / (float)cT;
    // D 16x16 f32 layout: lane -> N = lane%16, VGPR r -> M = half*8 + r
#pragma unroll
    for (int i = 0; i < 2; ++i) {
#pragma unroll
        for (int j = 0; j < 2; ++j) {
#pragma unroll
            for (int r = 0; r < 8; ++r) {
                const int l = l0 + i * 16 + half * 8 + r;
                const int m = m0 + j * 16 + lr;
                if (l < cL && m < cL)
                    sim[(((size_t)b * cS + s) * cL + l) * cL + m] = acc[i][j][r] * scale;
            }
        }
    }
}

// --- Kernel 2: softmax over joint (S*L) axis per (b, m); write bf16 attn ----
// Raw reshape (S*L,L)->(L,S*L) is identity on the flat buffer; we scatter into
// the padded layout attn_pad[b, l, j] (row pitch cSLP, pads pre-zeroed).
__global__ __launch_bounds__(256)
void softmax_kernel(const float* __restrict__ sim, __bf16* __restrict__ attn_pad) {
    const int m = blockIdx.x;   // 0..206
    const int b = blockIdx.y;
    const int tid = threadIdx.x;
    __shared__ float red[256];

    const float* col = sim + (size_t)b * cSL * cL + m;  // stride L over i

    float mx = -3.4e38f;
    for (int i = tid; i < cSL; i += 256) mx = fmaxf(mx, col[(size_t)i * cL]);
    red[tid] = mx; __syncthreads();
    for (int st = 128; st > 0; st >>= 1) {
        if (tid < st) red[tid] = fmaxf(red[tid], red[tid + st]);
        __syncthreads();
    }
    mx = red[0]; __syncthreads();

    float sum = 0.0f;
    for (int i = tid; i < cSL; i += 256) sum += __expf(col[(size_t)i * cL] - mx);
    red[tid] = sum; __syncthreads();
    for (int st = 128; st > 0; st >>= 1) {
        if (tid < st) red[tid] += red[tid + st];
        __syncthreads();
    }
    const float inv = 1.0f / red[0];

    for (int i = tid; i < cSL; i += 256) {
        const float v = __expf(col[(size_t)i * cL] - mx) * inv;
        const size_t f = (size_t)i * cL + m;        // flat index in [S*L, L]
        const int l = (int)(f / cSL);               // raw-reshape row
        const int j = (int)(f - (size_t)l * cSL);   // raw-reshape col (= s*L+link)
        attn_pad[((size_t)b * cL + l) * cSLP + j] = (__bf16)v;
    }
}

// --- Kernel 3: out[b,t,l,d] = sum_m attn[b,l,m] * x_flat[b,t,m,d] -----------
// One wave per (b, t, 32-row l tile); N = 64 -> 2x4 accumulators, B fragments
// (built by LDS column gather) shared across both A tiles. x_flat rows
// (m = s*L+link, data = x[b, t+p_s, link, :], zero if t+p_s >= T) are staged
// into LDS via gfx1250 async global->LDS copies when available.
__global__ __launch_bounds__(32)
void out_wmma_kernel(const __bf16* __restrict__ xbf,
                     const __bf16* __restrict__ attn_pad,
                     float* __restrict__ out) {
    const int lt = blockIdx.x;   // 0..6
    const int t  = blockIdx.y;   // 0..319
    const int b  = blockIdx.z;   // 0..1
    const int l0 = lt * 32;

    const int lane = threadIdx.x;
    const int lr   = lane & 15;
    const int half = lane >> 4;

    __shared__ __bf16 tile[32][72];   // [k_local][d], padded pitch (144B rows)

    const int lrow0 = (l0 + lr      < cL) ? (l0 + lr)      : (cL - 1);
    const int lrow1 = (l0 + 16 + lr < cL) ? (l0 + 16 + lr) : (cL - 1);
    const __bf16* arow0 = attn_pad + ((size_t)b * cL + lrow0) * cSLP;
    const __bf16* arow1 = attn_pad + ((size_t)b * cL + lrow1) * cSLP;

    v8f acc[2][4] = {};

    for (int kb = 0; kb < cSLP; kb += 32) {   // 136 K-chunks
        // ---- stage x_flat rows kb..kb+31 (32 bf16, d 0..31, of one row per lane) ----
        const int mm = kb + lane;
        const __bf16* src = nullptr;
        if (mm < cSL) {
            const int s    = mm / cL;
            const int link = mm - s * cL;
            const int tt   = t + shift_p(s);
            if (tt < cT)
                src = xbf + (((size_t)b * cT + tt) * cL + link) * cD;
        }
#if HAVE_ASYNC_LDS
        if (src) {
            AS1 v4i* g = (AS1 v4i*)(unsigned long long)src;
            AS3 v4i* l = (AS3 v4i*)(unsigned)(unsigned long long)&tile[lane][0];
            __builtin_amdgcn_global_load_async_to_lds_b128(g, l, 0, 0);
            __builtin_amdgcn_global_load_async_to_lds_b128(g, l, 16, 0);
            __builtin_amdgcn_global_load_async_to_lds_b128(g, l, 32, 0);
            __builtin_amdgcn_global_load_async_to_lds_b128(g, l, 48, 0);
        } else {
            const uint4 z = {};
            *(uint4*)&tile[lane][0]  = z;
            *(uint4*)&tile[lane][8]  = z;
            *(uint4*)&tile[lane][16] = z;
            *(uint4*)&tile[lane][24] = z;
        }
        __builtin_amdgcn_s_wait_asynccnt(0);
        __syncthreads();
#else
        {
            uint4 r0 = {}, r1 = {}, r2 = {}, r3 = {};
            if (src) {
                const uint4* rp = (const uint4*)src;
                r0 = rp[0]; r1 = rp[1]; r2 = rp[2]; r3 = rp[3];
            }
            *(uint4*)&tile[lane][0]  = r0;
            *(uint4*)&tile[lane][8]  = r1;
            *(uint4*)&tile[lane][16] = r2;
            *(uint4*)&tile[lane][24] = r3;
        }
        __syncthreads();
#endif

        // A fragments: attn rows are contiguous in m (16B-aligned thanks to cSLP)
        union U { v16bf v; uint4 u[2]; } a0, a1;
        a0.u[0] = *(const uint4*)(arow0 + kb + half * 8);
        a0.u[1] = *(const uint4*)(arow0 + kb + 16 + half * 8);
        a1.u[0] = *(const uint4*)(arow1 + kb + half * 8);
        a1.u[1] = *(const uint4*)(arow1 + kb + 16 + half * 8);

#pragma unroll
        for (int nt = 0; nt < 4; ++nt) {
            const int n  = nt * 16 + lr;   // d column
            const int k0 = half * 8;
            v16bf bv;
#pragma unroll
            for (int j = 0; j < 16; ++j) bv[j] = tile[(j < 8) ? (k0 + j) : (16 + k0 + j - 8)][n];
            acc[0][nt] = __builtin_amdgcn_wmma_f32_16x16x32_bf16(
                false, a0.v, false, bv, (short)0, acc[0][nt], false, false);
            acc[1][nt] = __builtin_amdgcn_wmma_f32_16x16x32_bf16(
                false, a1.v, false, bv, (short)0, acc[1][nt], false, false);
        }
        __syncthreads();
    }

#pragma unroll
    for (int i = 0; i < 2; ++i) {
#pragma unroll
        for (int nt = 0; nt < 4; ++nt) {
#pragma unroll
            for (int r = 0; r < 8; ++r) {
                const int l = l0 + i * 16 + half * 8 + r;
                const int d = nt * 16 + lr;
                if (l < cL)
                    out[(((size_t)b * cT + t) * cL + l) * cD + d] = acc[i][nt][r];
            }
        }
    }
}

// ---------------------------------------------------------------------------
extern "C" void kernel_launch(void* const* d_in, const int* in_sizes, int n_in,
                              void* d_out, int out_size, void* d_ws, size_t ws_size,
                              hipStream_t stream) {
    const float* x = (const float*)d_in[0];
    const float* w = (const float*)d_in[1];
    float* out = (float*)d_out;

    char* ws = (char*)d_ws;
    // workspace layout (all 256B aligned):
    //   xbf      : 8,478,720 bf16 = 16,957,440 B
    //   sim      : 1,799,658 f32  =  7,198,632 B
    //   attn_pad : 1,801,728 bf16 =  3,603,456 B        total ~27.8 MB
    __bf16* xbf      = (__bf16*)(ws);
    float*  sim      = (float*) (ws + 16957440);
    __bf16* attn_pad = (__bf16*)(ws + 24156160);

    const long long nconv = (long long)cB * cT * cL * cD;   // 8,478,720
    const long long npad  = (long long)cB * cL * cSLP;      // 1,801,728

    convert_zero_kernel<<<dim3((unsigned)((nconv + 255) / 256)), dim3(256), 0, stream>>>(
        x, xbf, attn_pad, nconv, npad);

    sim_wmma_kernel<<<dim3(cLT32 * cLT32, cS, cB), dim3(32), 0, stream>>>(xbf, w, sim);

    softmax_kernel<<<dim3(cL, cB), dim3(256), 0, stream>>>(sim, attn_pad);

    out_wmma_kernel<<<dim3(cLT32, cT, cB), dim3(32), 0, stream>>>(xbf, attn_pad, out);
}
